// GatedElementBasedRNNLayer_Loop_83408264888681
// MI455X (gfx1250) — compile-verified, matrix operand order checked by reading, not run on hardware
//
#include <hip/hip_runtime.h>
#include <hip/hip_bf16.h>
#include <math.h>

typedef __attribute__((ext_vector_type(16))) _Float16 v16h;
typedef __attribute__((ext_vector_type(8)))  float    v8f;

#define P_LEN 1000
#define Q_LEN 100
#define BATCH 128
#define D_IN  512
#define HID   256

// ---------------------------------------------------------------------------
// f32 -> f16 conversion (weights, done once)
// ---------------------------------------------------------------------------
__global__ void cvt_f32_to_f16(const float* __restrict__ src,
                               _Float16* __restrict__ dst, int n) {
  int i = blockIdx.x * blockDim.x + threadIdx.x;
  if (i < n) dst[i] = (_Float16)src[i];
}

__global__ void zero_f32(float* __restrict__ p, int n) {
  int i = blockIdx.x * blockDim.x + threadIdx.x;
  if (i < n) p[i] = 0.0f;
}

// ---------------------------------------------------------------------------
// WMMA GEMM tile body: one wave computes a 16x64 strip of
//   C[M,N] = A[M,K] (f32, cvt in-register) * B[N,K]^T (f16)
// Four 16x16 WMMA accumulators share one A fragment per K-step.
// MODE 0: C = acc ; MODE 1: C = acc + D ; MODE 2: C = D * sigmoid(acc)
// Requires M%16==0, N%64==0, K%32==0.
// ---------------------------------------------------------------------------
template <int MODE>
__device__ __forceinline__ void gemm_tile64(const float* __restrict__ A,
                                            const _Float16* __restrict__ B,
                                            const float* __restrict__ D,
                                            float* __restrict__ C,
                                            int M, int N, int K,
                                            int tile, int lane) {
  int ntn = N >> 6;
  int mt = tile / ntn;
  int nt = tile - mt * ntn;
  int hi = lane >> 4;     // K-half held by this lane
  int r  = lane & 15;     // row (A) / col (B) within the tile

  const float*    arow = A + (size_t)(mt * 16 + r) * K + hi * 16;
  const _Float16* brow = B + (size_t)(nt * 64 + r) * K + hi * 16;
  const size_t bstep = (size_t)16 * K;

  v8f acc0 = {}, acc1 = {}, acc2 = {}, acc3 = {};
  for (int k0 = 0; k0 < K; k0 += 32) {
    v16h a;
#pragma unroll
    for (int j = 0; j < 16; ++j) a[j] = (_Float16)arow[k0 + j];
    v16h b0 = *(const v16h*)(brow + k0);
    v16h b1 = *(const v16h*)(brow + bstep + k0);
    v16h b2 = *(const v16h*)(brow + 2 * bstep + k0);
    v16h b3 = *(const v16h*)(brow + 3 * bstep + k0);
    acc0 = __builtin_amdgcn_wmma_f32_16x16x32_f16(false, a, false, b0, (short)0, acc0, false, false);
    acc1 = __builtin_amdgcn_wmma_f32_16x16x32_f16(false, a, false, b1, (short)0, acc1, false, false);
    acc2 = __builtin_amdgcn_wmma_f32_16x16x32_f16(false, a, false, b2, (short)0, acc2, false, false);
    acc3 = __builtin_amdgcn_wmma_f32_16x16x32_f16(false, a, false, b3, (short)0, acc3, false, false);
  }

  v8f accs[4] = {acc0, acc1, acc2, acc3};
#pragma unroll
  for (int tt = 0; tt < 4; ++tt) {
    int col = nt * 64 + tt * 16 + r;
#pragma unroll
    for (int v = 0; v < 8; ++v) {
      int row = mt * 16 + v + 8 * hi;
      size_t idx = (size_t)row * N + col;
      float y = accs[tt][v];
      if (MODE == 1) {
        y += D[idx];
      } else if (MODE == 2) {
        y = D[idx] / (1.0f + expf(-y));   // D * sigmoid(acc)
      }
      C[idx] = y;
    }
  }
}

template <int MODE>
__global__ void wmma_gemm64(const float* __restrict__ A,
                            const _Float16* __restrict__ B,
                            const float* __restrict__ D,
                            float* __restrict__ C,
                            int M, int N, int K) {
  int wave = (int)((blockIdx.x * blockDim.x + threadIdx.x) >> 5);
  int lane = threadIdx.x & 31;
  int total = (M >> 4) * (N >> 6);
  if (wave >= total) return;
  gemm_tile64<MODE>(A, B, D, C, M, N, K, wave, lane);
}

// Two independent GEMMs with identical (M, N), different A/B/K, in one launch
// (used for the GRU gate pre-activations: G1 = x@W_ih^T, G2 = prev@W_hh^T).
__global__ void wmma_gemm64_dual(const float* __restrict__ A0,
                                 const _Float16* __restrict__ B0,
                                 float* __restrict__ C0, int K0,
                                 const float* __restrict__ A1,
                                 const _Float16* __restrict__ B1,
                                 float* __restrict__ C1, int K1,
                                 int M, int N) {
  int wave = (int)((blockIdx.x * blockDim.x + threadIdx.x) >> 5);
  int lane = threadIdx.x & 31;
  int t0 = (M >> 4) * (N >> 6);
  if (wave < t0)
    gemm_tile64<0>(A0, B0, nullptr, C0, M, N, K0, wave, lane);
  else if (wave < 2 * t0)
    gemm_tile64<0>(A1, B1, nullptr, C1, M, N, K1, wave - t0, lane);
}

// ---------------------------------------------------------------------------
// Fused attention for one batch b (block of 256 threads):
//   logits[q] = sum_h vT[h]*tanh(Qp[q,b,h] + u[b,h])     (wave per q, strided)
//   a = softmax_q(logits)
//   ct[h]    = sum_q a[q]*Qp[q,b,h]
// writes utct[b] = concat(Pp_i[b], ct)
// ---------------------------------------------------------------------------
__global__ void attn_fused(const float* __restrict__ Qp,
                           const float* __restrict__ u,
                           const float* __restrict__ vT,
                           const float* __restrict__ Ppi,
                           float* __restrict__ utct) {
  __shared__ float su[HID];
  __shared__ float sv[HID];
  __shared__ float lgt[Q_LEN];
  __shared__ float red[256];
  int b = blockIdx.x;
  int t = threadIdx.x;

  su[t] = u[(size_t)b * HID + t];
  sv[t] = vT[t];
  __syncthreads();

  int widx = t >> 5;
  int lane = t & 31;
  for (int q = widx; q < Q_LEN; q += 8) {
    const float* qp = Qp + ((size_t)q * BATCH + b) * HID;
    float acc = 0.0f;
#pragma unroll
    for (int h = lane; h < HID; h += 32)
      acc += sv[h] * tanhf(qp[h] + su[h]);
#pragma unroll
    for (int o = 16; o > 0; o >>= 1) acc += __shfl_xor(acc, o, 32);
    if (lane == 0) lgt[q] = acc;
  }
  __syncthreads();

  float l = (t < Q_LEN) ? lgt[t] : -3.4e38f;
  red[t] = l;
  __syncthreads();
  for (int s = 128; s > 0; s >>= 1) {
    if (t < s) red[t] = fmaxf(red[t], red[t + s]);
    __syncthreads();
  }
  float mx = red[0];
  __syncthreads();
  float e = (t < Q_LEN) ? expf(l - mx) : 0.0f;
  red[t] = e;
  __syncthreads();
  for (int s = 128; s > 0; s >>= 1) {
    if (t < s) red[t] += red[t + s];
    __syncthreads();
  }
  float inv = 1.0f / red[0];
  __syncthreads();
  if (t < Q_LEN) lgt[t] = e * inv;    // softmax weights
  __syncthreads();

  // thread t owns h = t (HID == blockDim.x)
  float ct = 0.0f;
  const float* qpb = Qp + (size_t)b * HID + t;
#pragma unroll 4
  for (int q = 0; q < Q_LEN; ++q)
    ct += lgt[q] * qpb[(size_t)q * BATCH * HID];

  utct[(size_t)b * (2 * HID) + t]       = Ppi[(size_t)b * HID + t];
  utct[(size_t)b * (2 * HID) + HID + t] = ct;
}

// ---------------------------------------------------------------------------
// GRU combine: r,z,n from G1 = x@W_ih^T, G2 = prev@W_hh^T (both [B,3H])
// h = (1-z)*n + z*prev ; writes prev (in place) and out[i]
// ---------------------------------------------------------------------------
__global__ void gru_combine(const float* __restrict__ G1,
                            const float* __restrict__ G2,
                            float* __restrict__ prev,
                            float* __restrict__ out_i) {
  int idx = blockIdx.x * blockDim.x + threadIdx.x;
  if (idx >= BATCH * HID) return;
  int b = idx / HID;
  int h = idx - b * HID;
  const float* g1 = G1 + (size_t)b * 3 * HID;
  const float* g2 = G2 + (size_t)b * 3 * HID;
  float r = 1.0f / (1.0f + expf(-(g1[h] + g2[h])));
  float z = 1.0f / (1.0f + expf(-(g1[HID + h] + g2[HID + h])));
  float n = tanhf(g1[2 * HID + h] + r * g2[2 * HID + h]);
  float p = prev[idx];
  float hn = (1.0f - z) * n + z * p;
  prev[idx] = hn;
  out_i[idx] = hn;
}

// ---------------------------------------------------------------------------
static inline void launch_gemm(int mode, const float* A, const _Float16* B,
                               const float* D, float* C, int M, int N, int K,
                               hipStream_t s) {
  int waves = (M >> 4) * (N >> 6);
  int blocks = (waves * 32 + 255) / 256;
  if (mode == 0)
    wmma_gemm64<0><<<blocks, 256, 0, s>>>(A, B, D, C, M, N, K);
  else if (mode == 1)
    wmma_gemm64<1><<<blocks, 256, 0, s>>>(A, B, D, C, M, N, K);
  else
    wmma_gemm64<2><<<blocks, 256, 0, s>>>(A, B, D, C, M, N, K);
}

extern "C" void kernel_launch(void* const* d_in, const int* in_sizes, int n_in,
                              void* d_out, int out_size, void* d_ws, size_t ws_size,
                              hipStream_t stream) {
  // Inputs in setup_inputs() order
  const float* passage  = (const float*)d_in[0];  // [P,B,D]
  const float* question = (const float*)d_in[1];  // [Q,B,D]
  // d_in[2], d_in[3]: masks (unused by the reference)
  const float* vT   = (const float*)d_in[4];      // [H]
  const float* WuQ  = (const float*)d_in[5];      // [H,D]
  const float* WuP  = (const float*)d_in[6];      // [H,D]
  const float* WvP  = (const float*)d_in[7];      // [H,H]
  const float* Wg   = (const float*)d_in[8];      // [2H,2H]
  const float* W_ih = (const float*)d_in[9];      // [3H,2H]
  const float* W_hh = (const float*)d_in[10];     // [3H,H]
  float* out = (float*)d_out;                     // [P,B,H]

  // ---- workspace carve-up -------------------------------------------------
  char* w = (char*)d_ws;
  auto carve = [&](size_t bytes) -> char* {
    char* p = w;
    w += (bytes + 255) & ~(size_t)255;
    return p;
  };
  float* Qp   = (float*)carve((size_t)Q_LEN * BATCH * HID * 4);   // 13.1 MB
  float* Pp   = (float*)carve((size_t)P_LEN * BATCH * HID * 4);   // 131 MB
  float* u    = (float*)carve((size_t)BATCH * HID * 4);
  float* utct = (float*)carve((size_t)BATCH * 2 * HID * 4);
  float* xg   = (float*)carve((size_t)BATCH * 2 * HID * 4);
  float* G1   = (float*)carve((size_t)BATCH * 3 * HID * 4);
  float* G2   = (float*)carve((size_t)BATCH * 3 * HID * 4);
  float* prev = (float*)carve((size_t)BATCH * HID * 4);
  _Float16* WuQh = (_Float16*)carve((size_t)HID * D_IN * 2);
  _Float16* WuPh = (_Float16*)carve((size_t)HID * D_IN * 2);
  _Float16* WvPh = (_Float16*)carve((size_t)HID * HID * 2);
  _Float16* Wgh  = (_Float16*)carve((size_t)2 * HID * 2 * HID * 2);
  _Float16* Wihh = (_Float16*)carve((size_t)3 * HID * 2 * HID * 2);
  _Float16* Whhh = (_Float16*)carve((size_t)3 * HID * HID * 2);
  (void)ws_size; (void)in_sizes; (void)n_in; (void)out_size;

  auto cvt = [&](const float* s, _Float16* d, int n) {
    cvt_f32_to_f16<<<(n + 255) / 256, 256, 0, stream>>>(s, d, n);
  };
  cvt(WuQ,  WuQh, HID * D_IN);
  cvt(WuP,  WuPh, HID * D_IN);
  cvt(WvP,  WvPh, HID * HID);
  cvt(Wg,   Wgh,  2 * HID * 2 * HID);
  cvt(W_ih, Wihh, 3 * HID * 2 * HID);
  cvt(W_hh, Whhh, 3 * HID * HID);

  zero_f32<<<(BATCH * HID + 255) / 256, 256, 0, stream>>>(prev, BATCH * HID);

  // Hoisted projections: Qp = question @ WuQ^T ; Pp = passage @ WuP^T
  launch_gemm(0, question, WuQh, nullptr, Qp, Q_LEN * BATCH, HID, D_IN, stream);
  launch_gemm(0, passage,  WuPh, nullptr, Pp, P_LEN * BATCH, HID, D_IN, stream);

  const int gru_waves  = 2 * (BATCH >> 4) * ((3 * HID) >> 6);   // dual GEMM
  const int gru_blocks = (gru_waves * 32 + 255) / 256;

  for (int i = 0; i < P_LEN; ++i) {
    const float* Ppi = Pp + (size_t)i * BATCH * HID;

    // u = Pp_i + prev @ WvP^T
    launch_gemm(1, prev, WvPh, Ppi, u, BATCH, HID, HID, stream);

    // fused attention: logits -> softmax -> context ; writes utct=[Pp_i, ct]
    attn_fused<<<BATCH, 256, 0, stream>>>(Qp, u, vT, Ppi, utct);

    // x = utct * sigmoid(utct @ Wg^T)
    launch_gemm(2, utct, Wgh, utct, xg, BATCH, 2 * HID, 2 * HID, stream);

    // GRU gate pre-activations: G1 = x@W_ih^T, G2 = prev@W_hh^T (one launch)
    wmma_gemm64_dual<<<gru_blocks, 256, 0, stream>>>(
        xg, Wihh, G1, 2 * HID, prev, Whhh, G2, HID, BATCH, 3 * HID);

    // combine -> h ; update prev and write out[i]
    gru_combine<<<(BATCH * HID + 255) / 256, 256, 0, stream>>>(
        G1, G2, prev, out + (size_t)i * BATCH * HID);
  }
}
